// FusedMoE_23596550324598
// MI455X (gfx1250) — matrix operand order repaired and checked by reading, not compiled, not run on hardware
//
#include <hip/hip_runtime.h>

#define T_TOKENS 1024
#define HIDDEN   4096
#define INTER    4096
#define NEXP     8

#define BM 64
#define BK 32
#define LSTRIDE 40   // padded LDS row stride in bf16 elements (80B rows -> conflict-free b128 lane loads)

typedef __attribute__((ext_vector_type(16))) __bf16 v16bf;
typedef __attribute__((ext_vector_type(8)))  __bf16 v8bf;
typedef __attribute__((ext_vector_type(8)))  float  v8f;
typedef __attribute__((ext_vector_type(4)))  int    v4i;

union FragBF { v16bf v; uint4 q[2]; };

#define AS1 __attribute__((address_space(1)))
#define AS3 __attribute__((address_space(3)))

#if defined(__AMDGCN__) && __has_builtin(__builtin_amdgcn_global_load_async_to_lds_b128) && __has_builtin(__builtin_amdgcn_s_wait_asynccnt)
#define USE_ASYNC_LDS 1
#else
#define USE_ASYNC_LDS 0
#endif

__device__ __forceinline__ void async_copy16(const void* g, void* l) {
#if USE_ASYNC_LDS
  __builtin_amdgcn_global_load_async_to_lds_b128((AS1 v4i*)g, (AS3 v4i*)l, 0, 0);
#else
  *(uint4*)l = *(const uint4*)g;
#endif
}
__device__ __forceinline__ void wait_async0() {
#if USE_ASYNC_LDS
  __builtin_amdgcn_s_wait_asynccnt(0);
#endif
}

// fp32 -> bf16 via hardware conversion (fptrunc is round-to-nearest-even)
__device__ __forceinline__ unsigned short f2bf(float f) {
  union { __bf16 h; unsigned short s; } u;
  u.h = (__bf16)f;
  return u.s;
}

__device__ __forceinline__ uint4 pack8(float4 a, float4 b) {
  union { v8bf v; uint4 q; } u;
  u.v = (v8bf){(__bf16)a.x, (__bf16)a.y, (__bf16)a.z, (__bf16)a.w,
               (__bf16)b.x, (__bf16)b.y, (__bf16)b.z, (__bf16)b.w};
  return u.q;
}

__device__ __forceinline__ v8f wmma_bf16(const FragBF& a, const FragBF& b, v8f c) {
  return __builtin_amdgcn_wmma_f32_16x16x32_bf16(false, a.v, false, b.v, (short)0, c, false, false);
}

__device__ __forceinline__ void load_frag_a(FragBF& a, const unsigned short (*As)[LSTRIDE], int fm, int fh) {
  a.q[0] = *(const uint4*)&As[fm][fh * 8];
  a.q[1] = *(const uint4*)&As[fm][16 + fh * 8];
}
__device__ __forceinline__ void load_frag_b(FragBF& b, const unsigned short (*Bs)[LSTRIDE], int n, int fkb) {
  b.q[0] = *(const uint4*)&Bs[n][fkb / 2];
  b.q[1] = *(const uint4*)&Bs[n][16 + fkb / 2];
}

__device__ __forceinline__ void g1_compute(
    const unsigned short (*as)[LSTRIDE],
    const unsigned short (*bg)[LSTRIDE],
    const unsigned short (*bu)[LSTRIDE],
    int fm, int fh, int fn, int fkb, int c0,
    v8f& aG0, v8f& aG1, v8f& aU0, v8f& aU1) {
  FragBF a; load_frag_a(a, as, fm, fh);
  FragBF bg0, bg1, bu0, bu1;
  load_frag_b(bg0, bg, c0 * 16 + fn, fkb);
  load_frag_b(bg1, bg, (c0 + 1) * 16 + fn, fkb);
  load_frag_b(bu0, bu, c0 * 16 + fn, fkb);
  load_frag_b(bu1, bu, (c0 + 1) * 16 + fn, fkb);
  aG0 = wmma_bf16(a, bg0, aG0);
  aG1 = wmma_bf16(a, bg1, aG1);
  aU0 = wmma_bf16(a, bu0, aU0);
  aU1 = wmma_bf16(a, bu1, aU1);
}

__device__ __forceinline__ void g2_compute(
    const unsigned short (*as)[LSTRIDE],
    const unsigned short (*bs)[LSTRIDE],
    int fm, int fh, int fn, int fkb, int ct0,
    v8f& acc0, v8f& acc1, v8f& acc2, v8f& acc3) {
  FragBF a; load_frag_a(a, as, fm, fh);
  FragBF f0, f1, f2, f3;
  load_frag_b(f0, bs, (ct0 + 0) * 16 + fn, fkb);
  load_frag_b(f1, bs, (ct0 + 1) * 16 + fn, fkb);
  load_frag_b(f2, bs, (ct0 + 2) * 16 + fn, fkb);
  load_frag_b(f3, bs, (ct0 + 3) * 16 + fn, fkb);
  acc0 = wmma_bf16(a, f0, acc0);
  acc1 = wmma_bf16(a, f1, acc1);
  acc2 = wmma_bf16(a, f2, acc2);
  acc3 = wmma_bf16(a, f3, acc3);
}

// ---------------- init: zero expert counts ----------------
__global__ void k_init(int* counts) {
  if (threadIdx.x < NEXP) counts[threadIdx.x] = 0;
}

// ---------------- convert hidden_states fp32 -> bf16 ----------------
__global__ void k_convert_x(const float* __restrict__ x, unsigned int* __restrict__ xb, int npairs) {
  int i = blockIdx.x * blockDim.x + threadIdx.x;
  if (i < npairs) {
    float2 v = ((const float2*)x)[i];
    xb[i] = (unsigned)f2bf(v.x) | ((unsigned)f2bf(v.y) << 16);
  }
}

// ---------------- routing: softmax + top2 + renorm + scatter ----------------
__global__ void k_route(const float* __restrict__ logits,
                        int* __restrict__ counts,
                        int* __restrict__ pairs,     // [E, T]
                        float* __restrict__ wgt) {   // [2T]
  int t = blockIdx.x * blockDim.x + threadIdx.x;
  if (t >= T_TOKENS) return;
  float l[NEXP];
  float mx = -3.0e38f;
  for (int e = 0; e < NEXP; ++e) { l[e] = logits[t * NEXP + e]; mx = fmaxf(mx, l[e]); }
  for (int e = 0; e < NEXP; ++e) l[e] = __expf(l[e] - mx);   // softmax denom cancels in renorm
  int e0 = 0; float p0 = l[0];
  for (int e = 1; e < NEXP; ++e) if (l[e] > p0) { p0 = l[e]; e0 = e; }
  int e1 = (e0 == 0) ? 1 : 0; float p1 = l[e1];
  for (int e = 0; e < NEXP; ++e) if (e != e0 && l[e] > p1) { p1 = l[e]; e1 = e; }
  float inv = 1.0f / (p0 + p1);
  int s0 = atomicAdd(&counts[e0], 1);
  pairs[e0 * T_TOKENS + s0] = 2 * t;
  wgt[2 * t] = p0 * inv;
  int s1 = atomicAdd(&counts[e1], 1);
  pairs[e1 * T_TOKENS + s1] = 2 * t + 1;
  wgt[2 * t + 1] = p1 * inv;
}

// ---------------- GEMM1: gathered X @ W13^T, fused SiLU(gate)*up -> Hbuf (bf16) ----------------
// 8 waves; tile BM=64 x (64 gate + 64 up); double-buffered LDS; async A copies; B regs pipelined 2 deep.
#define G1_BN 64
__global__ __launch_bounds__(256) void k_gemm1(
    const unsigned short* __restrict__ Xb,   // [T, H] bf16
    const float* __restrict__ W13,           // [E, 2I, H] fp32
    const int* __restrict__ counts,
    const int* __restrict__ pairs,           // [E, T]
    unsigned short* __restrict__ Hbuf) {     // [2T, I] bf16
  const int e   = blockIdx.z;
  const int cnt = counts[e];
  const int m0  = blockIdx.x * BM;
  if (m0 >= cnt) return;
  const int n0  = blockIdx.y * G1_BN;

  __shared__ int sPairs[BM];
  __shared__ __align__(16) unsigned short As[2][BM][LSTRIDE];
  __shared__ __align__(16) unsigned short Bg[2][G1_BN][LSTRIDE];
  __shared__ __align__(16) unsigned short Bu[2][G1_BN][LSTRIDE];

  const int tid = threadIdx.x;
  if (tid < BM) {
    int idx = m0 + tid;
    sPairs[tid] = (idx < cnt) ? pairs[e * T_TOKENS + idx] : -1;
  }
  __syncthreads();

  const int lane = tid & 31;
  const int wave = tid >> 5;
  const int wm   = wave & 3;   // row tile 0..3
  const int wn2  = wave >> 2;  // col group 0..1 -> tiles {2*wn2, 2*wn2+1}

  v8f aG0 = {}, aG1 = {}, aU0 = {}, aU1 = {};

  const int arow = tid >> 2, achunk = tid & 3;    // A: 64 rows x 4x(8 bf16)
  const int brow = tid >> 2, bq = tid & 3;        // B: 64 rows x 4x(8 f32)
  const float* W13e = W13 + (size_t)e * (2 * INTER) * HIDDEN;

  const int   aP    = sPairs[arow];
  const unsigned short* aSrc = (aP >= 0) ? (Xb + (size_t)(aP >> 1) * HIDDEN + achunk * 8) : nullptr;
  const float* gSrc = W13e + (size_t)(n0 + brow) * HIDDEN + bq * 8;
  const float* uSrc = W13e + (size_t)(INTER + n0 + brow) * HIDDEN + bq * 8;

  const int fm = wm * 16 + (lane & 15);
  const int fh = lane >> 4;
  const int fn = lane & 15;
  const int fkb = (lane >> 4) * 16;
  const int c0 = wn2 * 2;

  // zero-fill padding rows in both A buffers (their lanes skip async copies)
  if (aP < 0) {
    uint4 z = {0u, 0u, 0u, 0u};
    *(uint4*)&As[0][arow][achunk * 8] = z;
    *(uint4*)&As[1][arow][achunk * 8] = z;
  }
  // prologue: tile 0 -> buffers 0; issue B loads for tile 1 (consumed next iteration)
  if (aP >= 0) async_copy16(aSrc, &As[0][arow][achunk * 8]);
  {
    float4 g0 = *(const float4*)gSrc,  g1 = *(const float4*)(gSrc + 4);
    float4 u0 = *(const float4*)uSrc,  u1 = *(const float4*)(uSrc + 4);
    *(uint4*)&Bg[0][brow][bq * 8] = pack8(g0, g1);
    *(uint4*)&Bu[0][brow][bq * 8] = pack8(u0, u1);
  }
  float4 pg0 = *(const float4*)(gSrc + BK), pg1 = *(const float4*)(gSrc + BK + 4);
  float4 pu0 = *(const float4*)(uSrc + BK), pu1 = *(const float4*)(uSrc + BK + 4);
  wait_async0();
  __syncthreads();

  const int NK = HIDDEN / BK;
  int cur = 0;
  for (int kt = 0; kt < NK - 1; ++kt) {
    const int nxt = cur ^ 1;
    // async A copy for tile kt+1
    if (aP >= 0) async_copy16(aSrc + (kt + 1) * BK, &As[nxt][arow][achunk * 8]);
    // issue B loads for tile kt+2 (clamped; consumed next iteration)
    const int k2 = (kt + 2 < NK ? kt + 2 : NK - 1) * BK;
    float4 ng0 = *(const float4*)(gSrc + k2), ng1 = *(const float4*)(gSrc + k2 + 4);
    float4 nu0 = *(const float4*)(uSrc + k2), nu1 = *(const float4*)(uSrc + k2 + 4);
    // compute on current tile
    g1_compute(As[cur], Bg[cur], Bu[cur], fm, fh, fn, fkb, c0, aG0, aG1, aU0, aU1);
    // convert + store B for tile kt+1 (loaded one iteration ago -> no load stall)
    *(uint4*)&Bg[nxt][brow][bq * 8] = pack8(pg0, pg1);
    *(uint4*)&Bu[nxt][brow][bq * 8] = pack8(pu0, pu1);
    pg0 = ng0; pg1 = ng1; pu0 = nu0; pu1 = nu1;
    wait_async0();
    __syncthreads();
    cur = nxt;
  }
  g1_compute(As[cur], Bg[cur], Bu[cur], fm, fh, fn, fkb, c0, aG0, aG1, aU0, aU1);

  // epilogue: h = silu(gate) * up  -> Hbuf[p, j] (bf16)
  const int nl = lane & 15;
  const int mhalf = (lane >> 4) * 8;
  #pragma unroll
  for (int i = 0; i < 8; ++i) {
    int ml = wm * 16 + i + mhalf;
    int p = sPairs[ml];
    if (p < 0) continue;
    int j0 = n0 + c0 * 16 + nl;
    float g0 = aG0[i], u0 = aU0[i];
    float h0 = (g0 / (1.0f + __expf(-g0))) * u0;
    Hbuf[(size_t)p * INTER + j0] = f2bf(h0);
    float g1 = aG1[i], u1 = aU1[i];
    float h1 = (g1 / (1.0f + __expf(-g1))) * u1;
    Hbuf[(size_t)p * INTER + j0 + 16] = f2bf(h1);
  }
}

// ---------------- GEMM2: gathered H @ W2^T, scaled by routing weight -> pair_out (f32) ----------------
// 8 waves; tile BM=64 x BN=128; 4 accumulators/wave; same pipelining as GEMM1.
#define G2_BN 128
__global__ __launch_bounds__(256) void k_gemm2(
    const unsigned short* __restrict__ Hbuf, // [2T, I] bf16
    const float* __restrict__ W2,            // [E, H, I] fp32
    const int* __restrict__ counts,
    const int* __restrict__ pairs,
    const float* __restrict__ wgt,           // [2T]
    float* __restrict__ pout) {              // [2T, H] f32
  const int e   = blockIdx.z;
  const int cnt = counts[e];
  const int m0  = blockIdx.x * BM;
  if (m0 >= cnt) return;
  const int n0  = blockIdx.y * G2_BN;

  __shared__ int sPairs[BM];
  __shared__ __align__(16) unsigned short As[2][BM][LSTRIDE];
  __shared__ __align__(16) unsigned short Bs[2][G2_BN][LSTRIDE];

  const int tid = threadIdx.x;
  if (tid < BM) {
    int idx = m0 + tid;
    sPairs[tid] = (idx < cnt) ? pairs[e * T_TOKENS + idx] : -1;
  }
  __syncthreads();

  const int lane = tid & 31;
  const int wave = tid >> 5;
  const int wm   = wave & 3;   // row tile 0..3
  const int wn4  = wave >> 2;  // col group 0..1, 4 tiles each

  v8f acc0 = {}, acc1 = {}, acc2 = {}, acc3 = {};

  const int arow = tid >> 2, achunk = tid & 3;   // A: 64 rows x 4x(8 bf16)
  const int brow = tid >> 1, bhalf = tid & 1;    // B: 128 rows x 2x(16 f32)
  const float* W2e = W2 + (size_t)e * HIDDEN * INTER;

  const int   aP    = sPairs[arow];
  const unsigned short* aSrc = (aP >= 0) ? (Hbuf + (size_t)aP * INTER + achunk * 8) : nullptr;
  const float* bSrc = W2e + (size_t)(n0 + brow) * INTER + bhalf * 16;

  const int fm = wm * 16 + (lane & 15);
  const int fh = lane >> 4;
  const int fn = lane & 15;
  const int fkb = (lane >> 4) * 16;
  const int ct0 = wn4 * 4;

  if (aP < 0) {
    uint4 z = {0u, 0u, 0u, 0u};
    *(uint4*)&As[0][arow][achunk * 8] = z;
    *(uint4*)&As[1][arow][achunk * 8] = z;
  }
  if (aP >= 0) async_copy16(aSrc, &As[0][arow][achunk * 8]);
  {
    float4 b0 = *(const float4*)bSrc,       b1 = *(const float4*)(bSrc + 4);
    float4 b2 = *(const float4*)(bSrc + 8), b3 = *(const float4*)(bSrc + 12);
    *(uint4*)&Bs[0][brow][bhalf * 16]     = pack8(b0, b1);
    *(uint4*)&Bs[0][brow][bhalf * 16 + 8] = pack8(b2, b3);
  }
  float4 pb0 = *(const float4*)(bSrc + BK),     pb1 = *(const float4*)(bSrc + BK + 4);
  float4 pb2 = *(const float4*)(bSrc + BK + 8), pb3 = *(const float4*)(bSrc + BK + 12);
  wait_async0();
  __syncthreads();

  const int NK = INTER / BK;
  int cur = 0;
  for (int kt = 0; kt < NK - 1; ++kt) {
    const int nxt = cur ^ 1;
    if (aP >= 0) async_copy16(aSrc + (kt + 1) * BK, &As[nxt][arow][achunk * 8]);
    const int k2 = (kt + 2 < NK ? kt + 2 : NK - 1) * BK;
    float4 nb0 = *(const float4*)(bSrc + k2),     nb1 = *(const float4*)(bSrc + k2 + 4);
    float4 nb2 = *(const float4*)(bSrc + k2 + 8), nb3 = *(const float4*)(bSrc + k2 + 12);
    g2_compute(As[cur], Bs[cur], fm, fh, fn, fkb, ct0, acc0, acc1, acc2, acc3);
    *(uint4*)&Bs[nxt][brow][bhalf * 16]     = pack8(pb0, pb1);
    *(uint4*)&Bs[nxt][brow][bhalf * 16 + 8] = pack8(pb2, pb3);
    pb0 = nb0; pb1 = nb1; pb2 = nb2; pb3 = nb3;
    wait_async0();
    __syncthreads();
    cur = nxt;
  }
  g2_compute(As[cur], Bs[cur], fm, fh, fn, fkb, ct0, acc0, acc1, acc2, acc3);

  const int nl = lane & 15;
  const int mhalf = (lane >> 4) * 8;
  #pragma unroll
  for (int i = 0; i < 8; ++i) {
    int ml = wm * 16 + i + mhalf;
    int p = sPairs[ml];
    if (p < 0) continue;
    float w = wgt[p];
    float* dst = pout + (size_t)p * HIDDEN + n0 + ct0 * 16 + nl;
    dst[0]  = acc0[i] * w;
    dst[16] = acc1[i] * w;
    dst[32] = acc2[i] * w;
    dst[48] = acc3[i] * w;
  }
}

// ---------------- deterministic combine: out[t] = pout[2t] + pout[2t+1] ----------------
__global__ void k_combine(const float* __restrict__ pout, float* __restrict__ out, int n) {
  int i = blockIdx.x * blockDim.x + threadIdx.x;
  if (i < n) {
    int t = i >> 12;           // / HIDDEN
    int h = i & (HIDDEN - 1);
    out[i] = pout[(size_t)(2 * t) * HIDDEN + h] + pout[(size_t)(2 * t + 1) * HIDDEN + h];
  }
}

extern "C" void kernel_launch(void* const* d_in, const int* in_sizes, int n_in,
                              void* d_out, int out_size, void* d_ws, size_t ws_size,
                              hipStream_t stream) {
  (void)in_sizes; (void)n_in; (void)out_size; (void)ws_size;
  const float* hs  = (const float*)d_in[0];
  const float* rl  = (const float*)d_in[1];
  const float* w13 = (const float*)d_in[2];
  const float* w2  = (const float*)d_in[3];
  float* out = (float*)d_out;

  char* ws = (char*)d_ws;
  size_t off = 0;
  unsigned short* Xb   = (unsigned short*)(ws + off); off += (size_t)T_TOKENS * HIDDEN * 2;
  unsigned short* Hbuf = (unsigned short*)(ws + off); off += (size_t)2 * T_TOKENS * INTER * 2;
  float* pout  = (float*)(ws + off); off += (size_t)2 * T_TOKENS * HIDDEN * 4;
  float* wgt   = (float*)(ws + off); off += (size_t)2 * T_TOKENS * 4;
  int*   pairs = (int*)(ws + off);   off += (size_t)NEXP * T_TOKENS * 4;
  int*   counts= (int*)(ws + off);   off += 256;

  k_init<<<1, 32, 0, stream>>>(counts);
  int npairs = T_TOKENS * HIDDEN / 2;
  k_convert_x<<<(npairs + 255) / 256, 256, 0, stream>>>(hs, (unsigned int*)Xb, npairs);
  k_route<<<(T_TOKENS + 255) / 256, 256, 0, stream>>>(rl, counts, pairs, wgt);

  dim3 g1(T_TOKENS / BM, INTER / G1_BN, NEXP);   // m fastest -> L2 reuse of weight tiles
  k_gemm1<<<g1, 256, 0, stream>>>(Xb, w13, counts, pairs, Hbuf);

  dim3 g2(T_TOKENS / BM, HIDDEN / G2_BN, NEXP);
  k_gemm2<<<g2, 256, 0, stream>>>(Hbuf, w2, counts, pairs, wgt, pout);

  int n = T_TOKENS * HIDDEN;
  k_combine<<<(n + 255) / 256, 256, 0, stream>>>(pout, out, n);
}
